// DualBranchTracker2D_8976481649117
// MI455X (gfx1250) — compile-verified, matrix operand order checked by reading, not run on hardware
//
#include <hip/hip_runtime.h>
#include <hip/hip_bf16.h>
#include <math.h>

// ---------------- CDNA5 WMMA helpers (wave32, 16x16x32 f16 -> f32) ----------
typedef __attribute__((ext_vector_type(16))) _Float16 v16h;
typedef __attribute__((ext_vector_type(8)))  _Float16 v8h;
typedef __attribute__((ext_vector_type(8)))  float    v8f;

#define NB   64      // B
#define NT   512     // T
#define NC   16      // C
#define NF   16      // F
#define NG   8       // G
#define ND   32      // D
#define NGH  128     // GRU hidden
#define NS   (NB*NT) // 32768 rows
#define N2   (NS*2)  // 65536 per-output elems

#define OFF_PHAT  0
#define OFF_PABS  (N2)
#define OFF_PDYN  (2*N2)
#define OFF_V     (3*N2)
#define OFF_ALPHA (4*N2)

// Packed B-fragment arena tile offsets (1 tile = 32 lanes x 16 halfs = 512 h)
#define P_WP    0     // ch_proj_w  (32,24)   2 tiles  [nt]
#define P_WQKV  2     // qkv_w      (96,32)   6 tiles  [nt]
#define P_WO    8     // out_w      (32,32)   2 tiles  [nt]
#define P_WF1   10    // ff1_w      (128,32)  8 tiles  [nt]
#define P_WF2   18    // ff2_w      (32,128)  8 tiles  [nt*4+kt]
#define P_VW1   26    // vel_w1     (128,128) 32 tiles [nt*4+kt]
#define P_AW1   58    // abs_w1     (128,168) 48 tiles [nt*6+kt]
#define P_LW1   106   // al_w1      (64,130)  20 tiles [nt*5+kt]
#define P_WIH   126   // gru_wih    (384,64)  48 tiles [nt*2+kt]
#define P_WHH   174   // gru_whh    (384,128) 96 tiles [nt*4+kt]
#define P_TILES 270

__device__ __forceinline__ v8f wmma(v16h a, v16h b, v8f c) {
  return __builtin_amdgcn_wmma_f32_16x16x32_f16(false, a, false, b, (short)0, c,
                                                false, false);
}

// A fragment (16x32 f16) from row-major f16 with leading dim ld (ld % 8 == 0).
// lane<16: M=lane, K={0..7,16..23}; lane>=16: M=lane-16, K={8..15,24..31}.
// Per lane this is two contiguous 16-byte runs -> two b128 loads.
__device__ __forceinline__ v16h load_a16(const _Float16* base, int ld, int lane) {
  int row = lane & 15, hi = (lane >> 4) & 1;
  const _Float16* p = base + row * ld + hi * 8;
  v8h lo = *(const v8h*)(p);
  v8h hh = *(const v8h*)(p + 16);
  v16h a;
#pragma unroll
  for (int e = 0; e < 8; ++e) { a[e] = lo[e]; a[e + 8] = hh[e]; }
  return a;
}

// B fragment from packed arena: one 32-byte contiguous load per lane.
__device__ __forceinline__ v16h load_bp(const _Float16* pk, int tile, int lane) {
  return *(const v16h*)(pk + (size_t)tile * 512 + lane * 16);
}

__device__ __forceinline__ float sigmf(float x) { return 1.0f / (1.0f + __expf(-x)); }

// ---------------- Kernel: pack W [N][K] f32 -> B-fragment-ordered f16 -------
// outp[((nt*ktiles+kt)*32 + lane)*16 + e]; zero-padded outside (N,K).
__global__ __launch_bounds__(512) void k_pack_b(const float* __restrict__ w,
    _Float16* __restrict__ outp, int ktiles, int N, int K)
{
  int t = blockIdx.x;                    // tile id = nt*ktiles + kt
  int nt = t / ktiles, kt = t % ktiles;
  int lane = threadIdx.x >> 4;           // 0..31
  int e = threadIdx.x & 15;
  int hi = (lane >> 4) & 1;
  int k = kt * 32 + e + (e & 8) + hi * 8;
  int n = nt * 16 + (lane & 15);
  float v = (k < K && n < N) ? w[n * K + k] : 0.f;
  outp[((size_t)t * 32 + lane) * 16 + e] = (_Float16)v;
}

// ---------------- Kernel 1: ch_proj + transformer layer + mean pool ---------
// One wave per (b,t) sequence (16 tokens x 32 dims). 4 waves / block.
__global__ __launch_bounds__(128) void k_encoder(
    const float* __restrict__ cf, const unsigned char* __restrict__ pad,
    const float* __restrict__ gfeat, const _Float16* __restrict__ pk,
    const float* __restrict__ bp,  const float* __restrict__ bqkv,
    const float* __restrict__ bo,
    const float* __restrict__ ln1w,const float* __restrict__ ln1b,
    const float* __restrict__ bf1, const float* __restrict__ bf2,
    const float* __restrict__ ln2w,const float* __restrict__ ln2b,
    float* __restrict__ pool)
{
  __shared__ __align__(32) _Float16 SB[4][16 * 32];   // f16 staging (A source)
  __shared__ __align__(32) _Float16 FB[4][16 * 128];  // FF hidden (gelu), f16
  __shared__ float    ZB[4][16 * 32];                 // residual / z, f32
  __shared__ float    QB[4][16 * 96];                 // q,k,v f32

  int wave = threadIdx.x >> 5, lane = threadIdx.x & 31;
  int s = blockIdx.x * 4 + wave;        // sequence id in [0, B*T)
  _Float16* sb = SB[wave];
  _Float16* fb = FB[wave];
  float*    Z  = ZB[wave];
  float*    QKV= QB[wave];
  int col = lane & 15, hi = (lane >> 4) & 1;

  // ---- build X = [ch_feats | g | 0pad] as f16 [16 x 32] ----
  {
    const float* cfb = cf + (size_t)s * NC * NF;
    const float* gb_ = gfeat + (size_t)s * NG;
#pragma unroll
    for (int i = 0; i < 16; ++i) {
      int lin = lane * 16 + i;          // 0..511
      int c = lin >> 5, k = lin & 31;
      float v = (k < NF) ? cfb[c * NF + k]
              : (k < NF + NG) ? gb_[k - NF] : 0.0f;
      sb[c * 32 + k] = (_Float16)v;
    }
  }
  // ---- u = X @ Wp^T + bp  (K=24 padded to 32) ----
  {
    v16h a = load_a16(sb, 32, lane);
    v8f c0 = {}, c1 = {};
    c0 = wmma(a, load_bp(pk, P_WP + 0, lane), c0);
    c1 = wmma(a, load_bp(pk, P_WP + 1, lane), c1);
#pragma unroll
    for (int r = 0; r < 8; ++r) {
      int row = r + 8 * hi;
      float u0 = c0[r] + bp[col], u1 = c1[r] + bp[16 + col];
      Z[row * 32 + col] = u0;            Z[row * 32 + 16 + col] = u1;
      sb[row * 32 + col] = (_Float16)u0; sb[row * 32 + 16 + col] = (_Float16)u1;
    }
  }
  // ---- qkv = u @ Wqkv^T + b -> QKV [16 x 96] f32 ----
  {
    v16h a = load_a16(sb, 32, lane);
#pragma unroll
    for (int nt = 0; nt < 6; ++nt) {
      v8f c = {};
      c = wmma(a, load_bp(pk, P_WQKV + nt, lane), c);
#pragma unroll
      for (int r = 0; r < 8; ++r)
        QKV[(r + 8 * hi) * 96 + nt * 16 + col] = c[r] + bqkv[nt * 16 + col];
    }
  }
  // ---- attention (4 heads, dh=8), VALU; ao -> sb f16 ----
  {
    const unsigned char* pd = pad + (size_t)s * NC;
#pragma unroll
    for (int p = 0; p < 2; ++p) {
      int idx = p * 32 + lane;          // 0..63 -> (q_tok, head)
      int qt = idx & 15, h = idx >> 4;
      float sc[16], m = -1e30f;
      for (int k2 = 0; k2 < 16; ++k2) {
        float acc = 0.f;
#pragma unroll
        for (int d = 0; d < 8; ++d)
          acc += QKV[qt * 96 + h * 8 + d] * QKV[k2 * 96 + 32 + h * 8 + d];
        acc *= 0.35355339059f;
        if (pd[k2]) acc = -1e9f;
        sc[k2] = acc; m = fmaxf(m, acc);
      }
      float ssum = 0.f;
      for (int k2 = 0; k2 < 16; ++k2) { sc[k2] = __expf(sc[k2] - m); ssum += sc[k2]; }
      float inv = 1.f / ssum;
#pragma unroll
      for (int d = 0; d < 8; ++d) {
        float acc = 0.f;
        for (int k2 = 0; k2 < 16; ++k2)
          acc += sc[k2] * QKV[k2 * 96 + 64 + h * 8 + d];
        sb[qt * 32 + h * 8 + d] = (_Float16)(acc * inv);
      }
    }
  }
  // ---- out proj + residual ----
  {
    v16h a = load_a16(sb, 32, lane);
    v8f c0 = {}, c1 = {};
    c0 = wmma(a, load_bp(pk, P_WO + 0, lane), c0);
    c1 = wmma(a, load_bp(pk, P_WO + 1, lane), c1);
#pragma unroll
    for (int r = 0; r < 8; ++r) {
      int row = r + 8 * hi;
      Z[row * 32 + col]      += c0[r] + bo[col];
      Z[row * 32 + 16 + col] += c1[r] + bo[16 + col];
    }
  }
  // ---- LN1 (lanes 0..15, one token each); writes Z (f32) and sb (f16) ----
  if (lane < 16) {
    float mu = 0.f;
    for (int j = 0; j < 32; ++j) mu += Z[lane * 32 + j];
    mu *= (1.f / 32.f);
    float var = 0.f;
    for (int j = 0; j < 32; ++j) { float d = Z[lane * 32 + j] - mu; var += d * d; }
    float rstd = rsqrtf(var * (1.f / 32.f) + 1e-5f);
    for (int j = 0; j < 32; ++j) {
      float zz = (Z[lane * 32 + j] - mu) * rstd * ln1w[j] + ln1b[j];
      Z[lane * 32 + j] = zz; sb[lane * 32 + j] = (_Float16)zz;
    }
  }
  // ---- FF1 + exact gelu -> fb [16 x 128] f16 ----
  {
    v16h a = load_a16(sb, 32, lane);
#pragma unroll
    for (int nt = 0; nt < 8; ++nt) {
      v8f c = {};
      c = wmma(a, load_bp(pk, P_WF1 + nt, lane), c);
#pragma unroll
      for (int r = 0; r < 8; ++r) {
        float x = c[r] + bf1[nt * 16 + col];
        float gl = 0.5f * x * (1.f + erff(x * 0.70710678f));
        fb[(r + 8 * hi) * 128 + nt * 16 + col] = (_Float16)gl;
      }
    }
  }
  // ---- FF2 (K=128: 4 k-steps) + residual ----
  {
    v8f c0 = {}, c1 = {};
#pragma unroll
    for (int kt = 0; kt < 4; ++kt) {
      v16h a = load_a16(fb + kt * 32, 128, lane);
      c0 = wmma(a, load_bp(pk, P_WF2 + 0 * 4 + kt, lane), c0);
      c1 = wmma(a, load_bp(pk, P_WF2 + 1 * 4 + kt, lane), c1);
    }
#pragma unroll
    for (int r = 0; r < 8; ++r) {
      int row = r + 8 * hi;
      Z[row * 32 + col]      += c0[r] + bf2[col];
      Z[row * 32 + 16 + col] += c1[r] + bf2[16 + col];
    }
  }
  // ---- LN2 ----
  if (lane < 16) {
    float mu = 0.f;
    for (int j = 0; j < 32; ++j) mu += Z[lane * 32 + j];
    mu *= (1.f / 32.f);
    float var = 0.f;
    for (int j = 0; j < 32; ++j) { float d = Z[lane * 32 + j] - mu; var += d * d; }
    float rstd = rsqrtf(var * (1.f / 32.f) + 1e-5f);
    for (int j = 0; j < 32; ++j)
      Z[lane * 32 + j] = (Z[lane * 32 + j] - mu) * rstd * ln2w[j] + ln2b[j];
  }
  // ---- mean pool over 16 tokens (lane = channel 0..31) ----
  {
    float acc = 0.f;
    for (int tk = 0; tk < 16; ++tk) acc += Z[tk * 32 + lane];
    pool[(size_t)s * 32 + lane] = acc * (1.f / 16.f);
  }
}

// ---------------- Kernel: gin = [pool | g @ genc^T + b] as f16 --------------
__global__ void k_gin(const float* __restrict__ pool, const float* __restrict__ g,
                      const float* __restrict__ gw, const float* __restrict__ gb,
                      _Float16* __restrict__ gin16)
{
  int t = blockIdx.x * blockDim.x + threadIdx.x;
  if (t >= NS * 32) return;
  int s = t >> 5, j = t & 31;
  gin16[(size_t)s * 64 + j] = (_Float16)pool[(size_t)s * 32 + j];
  float acc = gb[j];
#pragma unroll
  for (int i = 0; i < 8; ++i) acc += g[(size_t)s * 8 + i] * gw[j * 8 + i];
  gin16[(size_t)s * 64 + 32 + j] = (_Float16)acc;
}

// ---------------- Kernel: gx = gin @ Wih^T + bih  [32768 x 384] -------------
__global__ __launch_bounds__(256) void k_gx(const _Float16* __restrict__ gin16,
    const _Float16* __restrict__ pk, const float* __restrict__ bih,
    float* __restrict__ gx)
{
  int gw = (blockIdx.x * 256 + threadIdx.x) >> 5;
  int lane = threadIdx.x & 31;
  int rt = gw / 24, nt = gw % 24;
  if (rt >= NS / 16) return;
  const _Float16* Ab = gin16 + (size_t)rt * 16 * 64;
  v8f c = {};
#pragma unroll
  for (int kt = 0; kt < 2; ++kt) {
    v16h a = load_a16(Ab + kt * 32, 64, lane);
    c = wmma(a, load_bp(pk, P_WIH + nt * 2 + kt, lane), c);
  }
  int col = lane & 15, hi = (lane >> 4) & 1;
#pragma unroll
  for (int r = 0; r < 8; ++r) {
    int row = rt * 16 + r + 8 * hi;
    gx[(size_t)row * 384 + nt * 16 + col] = c[r] + bih[nt * 16 + col];
  }
}

// ---------------- Kernel: persistent single-WGP GRU scan --------------------
// 32 waves: wave = (mt:4)x(jg:8). Wave (mt,jg) computes r/z/n gate tiles for
// batch rows mt*16..+15 and hidden cols jg*16..+15; h kept resident in LDS.
__global__ __launch_bounds__(1024) void k_gru(const float* __restrict__ gx,
    const _Float16* __restrict__ pk, const float* __restrict__ bhh,
    float* __restrict__ hseq)
{
  __shared__ float    hcur[NB * NGH];                  // 32 KB
  __shared__ __align__(32) _Float16 h16[NB * NGH];     // 16 KB
  int wave = threadIdx.x >> 5, lane = threadIdx.x & 31;
  int mt = wave >> 3, jg = wave & 7;
  for (int i = threadIdx.x; i < NB * NGH; i += 1024) { hcur[i] = 0.f; h16[i] = (_Float16)0.f; }
  __syncthreads();
  int col = lane & 15, hi = (lane >> 4) & 1;
  int n_r = jg * 16 + col;               // hidden column
  float bh_r = bhh[n_r], bh_z = bhh[128 + n_r], bh_n = bhh[256 + n_r];
  for (int t = 0; t < NT; ++t) {
    v8f cr = {}, cz = {}, cn = {};
#pragma unroll
    for (int kt = 0; kt < 4; ++kt) {
      v16h a = load_a16(h16 + (size_t)mt * 16 * NGH + kt * 32, NGH, lane);
      cr = wmma(a, load_bp(pk, P_WHH + (jg     ) * 4 + kt, lane), cr);
      cz = wmma(a, load_bp(pk, P_WHH + (jg +  8) * 4 + kt, lane), cz);
      cn = wmma(a, load_bp(pk, P_WHH + (jg + 16) * 4 + kt, lane), cn);
    }
    __syncthreads();                      // all reads of h16 done
    float h2v[8];
#pragma unroll
    for (int r = 0; r < 8; ++r) {
      int row = mt * 16 + r + 8 * hi;     // batch index
      const float* gxr = gx + ((size_t)row * NT + t) * 384;
      float rg = sigmf(gxr[n_r]       + cr[r] + bh_r);
      float zg = sigmf(gxr[128 + n_r] + cz[r] + bh_z);
      float ng = tanhf(gxr[256 + n_r] + rg * (cn[r] + bh_n));
      float hprev = hcur[row * NGH + n_r];
      float h2 = (1.f - zg) * ng + zg * hprev;
      h2v[r] = h2;
      hseq[((size_t)row * NT + t) * NGH + n_r] = h2;
    }
    __syncthreads();                      // hprev reads done before overwrite
#pragma unroll
    for (int r = 0; r < 8; ++r) {
      int row = mt * 16 + r + 8 * hi;
      hcur[row * NGH + n_r] = h2v[r];
      h16 [row * NGH + n_r] = (_Float16)h2v[r];
    }
    __syncthreads();                      // h updates visible for next step
  }
}

// ---------------- Kernel: per-(b,t) channel means for alpha head ------------
__global__ void k_means(const float* __restrict__ cf, float* __restrict__ mc,
                        float* __restrict__ mo)
{
  int s = blockIdx.x * blockDim.x + threadIdx.x;
  if (s >= NS) return;
  const float* base = cf + (size_t)s * NC * NF;
  float a = 0.f, b = 0.f;
#pragma unroll
  for (int c = 0; c < NC; ++c) {
    a += base[c * NF + 8];
    b += fmaxf(base[c * NF + 3], base[c * NF + 4]);
  }
  mc[s] = a * (1.f / 16.f); mo[s] = b * (1.f / 16.f);
}

// ---------------- Kernel: heads (vel / abs / alpha), wave per 16 rows -------
__global__ __launch_bounds__(128) void k_heads(
    const float* __restrict__ hseq, const float* __restrict__ pool,
    const float* __restrict__ gfeat, const float* __restrict__ mc,
    const float* __restrict__ mo, const _Float16* __restrict__ pk,
    const float* __restrict__ ab1, const float* __restrict__ aw2,
    const float* __restrict__ ab2,
    const float* __restrict__ vb1, const float* __restrict__ vw2,
    const float* __restrict__ vb2,
    const float* __restrict__ lb1, const float* __restrict__ lw2,
    const float* __restrict__ lb2,
    float* __restrict__ out)
{
  __shared__ __align__(32) _Float16 HB[4][16 * 128];
  __shared__ __align__(32) _Float16 CB[4][16 * 192];
  __shared__ __align__(32) _Float16 TB[4][16 * 128];
  int wave = threadIdx.x >> 5, lane = threadIdx.x & 31;
  int rt = blockIdx.x * 4 + wave;
  _Float16* hb = HB[wave]; _Float16* cb = CB[wave]; _Float16* tb = TB[wave];
  int row0 = rt * 16;
  int col = lane & 15, hi = (lane >> 4) & 1;
  for (int i = lane; i < 16 * 128; i += 32) {
    int rr = i >> 7, cc = i & 127;
    hb[i] = (_Float16)hseq[(size_t)(row0 + rr) * NGH + cc];
  }
  // ---- vel: relu(h @ vw1^T + b) @ vw2^T; v = 0.08*tanh(.) ----
#pragma unroll
  for (int nt = 0; nt < 8; ++nt) {
    v8f c = {};
#pragma unroll
    for (int kt = 0; kt < 4; ++kt) {
      v16h a = load_a16(hb + kt * 32, 128, lane);
      c = wmma(a, load_bp(pk, P_VW1 + nt * 4 + kt, lane), c);
    }
#pragma unroll
    for (int r = 0; r < 8; ++r)
      tb[(r + 8 * hi) * 128 + nt * 16 + col] = (_Float16)fmaxf(c[r] + vb1[nt * 16 + col], 0.f);
  }
  {
    int rr = lane & 15, o = (lane >> 4) & 1;
    float acc = vb2[o];
    for (int k = 0; k < 128; ++k) acc += (float)tb[rr * 128 + k] * vw2[o * 128 + k];
    out[OFF_V + ((size_t)(row0 + rr)) * 2 + o] = 0.08f * tanhf(acc);
  }
  // ---- abs: ctx=[h|pool|g] (K=168 pad 192) ----
  for (int i = lane; i < 16 * 192; i += 32) {
    int rr = i / 192, cc = i % 192;
    float v;
    if (cc < 128) v = (float)hb[rr * 128 + cc];
    else if (cc < 160) v = pool[(size_t)(row0 + rr) * 32 + (cc - 128)];
    else if (cc < 168) v = gfeat[(size_t)(row0 + rr) * 8 + (cc - 160)];
    else v = 0.f;
    cb[i] = (_Float16)v;
  }
#pragma unroll
  for (int nt = 0; nt < 8; ++nt) {
    v8f c = {};
#pragma unroll
    for (int kt = 0; kt < 6; ++kt) {
      v16h a = load_a16(cb + kt * 32, 192, lane);
      c = wmma(a, load_bp(pk, P_AW1 + nt * 6 + kt, lane), c);
    }
#pragma unroll
    for (int r = 0; r < 8; ++r)
      tb[(r + 8 * hi) * 128 + nt * 16 + col] = (_Float16)fmaxf(c[r] + ab1[nt * 16 + col], 0.f);
  }
  {
    int rr = lane & 15, o = (lane >> 4) & 1;
    float acc = ab2[o];
    for (int k = 0; k < 128; ++k) acc += (float)tb[rr * 128 + k] * aw2[o * 128 + k];
    out[OFF_PABS + ((size_t)(row0 + rr)) * 2 + o] = acc;
  }
  // ---- alpha: ain=[h|mean_c|mean_obs] (K=130 pad 160), N=64 ----
  for (int i = lane; i < 16 * 160; i += 32) {
    int rr = i / 160, cc = i % 160;
    float v;
    if (cc < 128) v = (float)hb[rr * 128 + cc];
    else if (cc == 128) v = mc[row0 + rr];
    else if (cc == 129) v = mo[row0 + rr];
    else v = 0.f;
    cb[i] = (_Float16)v;
  }
#pragma unroll
  for (int nt = 0; nt < 4; ++nt) {
    v8f c = {};
#pragma unroll
    for (int kt = 0; kt < 5; ++kt) {
      v16h a = load_a16(cb + kt * 32, 160, lane);
      c = wmma(a, load_bp(pk, P_LW1 + nt * 5 + kt, lane), c);
    }
#pragma unroll
    for (int r = 0; r < 8; ++r)
      tb[(r + 8 * hi) * 128 + nt * 16 + col] = (_Float16)fmaxf(c[r] + lb1[nt * 16 + col], 0.f);
  }
  {
    int rr = lane & 15, o = (lane >> 4) & 1;
    float acc = lb2[o];
    for (int k = 0; k < 64; ++k) acc += (float)tb[rr * 128 + k] * lw2[o * 64 + k];
    out[OFF_ALPHA + ((size_t)(row0 + rr)) * 2 + o] = sigmf(acc);
  }
}

// ---------------- Kernel: sequential rollout + blend ------------------------
__global__ void k_rollout(const float* __restrict__ gfeat,
                          const float* __restrict__ xy0, float* __restrict__ out)
{
  int b = threadIdx.x >> 1, d = threadIdx.x & 1;
  if (b >= NB) return;
  const float* v  = out + OFF_V;
  const float* pa = out + OFF_PABS;
  const float* al = out + OFF_ALPHA;
  float p = xy0[b * 2 + d];
  {
    float a0 = al[((size_t)b * NT) * 2 + d];
    out[OFF_PDYN + ((size_t)b * NT) * 2 + d] = p;
    out[OFF_PHAT + ((size_t)b * NT) * 2 + d] = a0 * pa[((size_t)b * NT) * 2 + d] + (1.f - a0) * p;
  }
  for (int t = 1; t < NT; ++t) {
    float dt = fmaxf(gfeat[((size_t)b * NT + t) * NG + 0], 1e-4f);
    p += v[((size_t)b * NT + (t - 1)) * 2 + d] * dt;   // forward == cumsum
    size_t i = ((size_t)b * NT + t) * 2 + d;
    float a = al[i];
    out[OFF_PDYN + i] = p;
    out[OFF_PHAT + i] = a * pa[i] + (1.f - a) * p;
  }
}

// ---------------- launch ----------------------------------------------------
extern "C" void kernel_launch(void* const* d_in, const int* in_sizes, int n_in,
                              void* d_out, int out_size, void* d_ws, size_t ws_size,
                              hipStream_t stream) {
  (void)in_sizes; (void)n_in; (void)out_size; (void)ws_size;
  const float* cf  = (const float*)d_in[0];
  const unsigned char* pad = (const unsigned char*)d_in[1];
  const float* g   = (const float*)d_in[2];
  const float* xy0 = (const float*)d_in[3];
  const float* wp  = (const float*)d_in[4];   const float* bp  = (const float*)d_in[5];
  const float* wqkv= (const float*)d_in[6];   const float* bqkv= (const float*)d_in[7];
  const float* wo  = (const float*)d_in[8];   const float* bo  = (const float*)d_in[9];
  const float* ln1w= (const float*)d_in[10];  const float* ln1b= (const float*)d_in[11];
  const float* wf1 = (const float*)d_in[12];  const float* bf1 = (const float*)d_in[13];
  const float* wf2 = (const float*)d_in[14];  const float* bf2 = (const float*)d_in[15];
  const float* ln2w= (const float*)d_in[16];  const float* ln2b= (const float*)d_in[17];
  const float* gw  = (const float*)d_in[18];  const float* gbi = (const float*)d_in[19];
  const float* wih = (const float*)d_in[20];  const float* whh = (const float*)d_in[21];
  const float* bih = (const float*)d_in[22];  const float* bhh = (const float*)d_in[23];
  const float* aw1 = (const float*)d_in[24];  const float* ab1 = (const float*)d_in[25];
  const float* aw2 = (const float*)d_in[26];  const float* ab2 = (const float*)d_in[27];
  const float* vw1 = (const float*)d_in[28];  const float* vb1 = (const float*)d_in[29];
  const float* vw2 = (const float*)d_in[30];  const float* vb2 = (const float*)d_in[31];
  const float* lw1 = (const float*)d_in[32];  const float* lb1 = (const float*)d_in[33];
  const float* lw2 = (const float*)d_in[34];  const float* lb2 = (const float*)d_in[35];
  float* out = (float*)d_out;

  // workspace layout (all offsets keep >=32B alignment)
  float* pool = (float*)d_ws;                                  // 32768*32
  float* gx   = pool + (size_t)NS * 32;                        // 32768*384
  float* hseq = gx   + (size_t)NS * 384;                       // 32768*128
  float* mc   = hseq + (size_t)NS * 128;                       // 32768
  float* mo   = mc + NS;                                       // 32768
  _Float16* gin16 = (_Float16*)(mo + NS);                      // 32768*64
  _Float16* pk    = gin16 + (size_t)NS * 64;                   // 270 tiles * 512 h

  // ---- pack all WMMA B operands into fragment-ordered f16 tiles ----
  k_pack_b<<<2,  512, 0, stream>>>(wp,   pk + (size_t)P_WP   * 512, 1, 32, 24);
  k_pack_b<<<6,  512, 0, stream>>>(wqkv, pk + (size_t)P_WQKV * 512, 1, 96, 32);
  k_pack_b<<<2,  512, 0, stream>>>(wo,   pk + (size_t)P_WO   * 512, 1, 32, 32);
  k_pack_b<<<8,  512, 0, stream>>>(wf1,  pk + (size_t)P_WF1  * 512, 1, 128, 32);
  k_pack_b<<<8,  512, 0, stream>>>(wf2,  pk + (size_t)P_WF2  * 512, 4, 32, 128);
  k_pack_b<<<32, 512, 0, stream>>>(vw1,  pk + (size_t)P_VW1  * 512, 4, 128, 128);
  k_pack_b<<<48, 512, 0, stream>>>(aw1,  pk + (size_t)P_AW1  * 512, 6, 128, 168);
  k_pack_b<<<20, 512, 0, stream>>>(lw1,  pk + (size_t)P_LW1  * 512, 5, 64, 130);
  k_pack_b<<<48, 512, 0, stream>>>(wih,  pk + (size_t)P_WIH  * 512, 2, 384, 64);
  k_pack_b<<<96, 512, 0, stream>>>(whh,  pk + (size_t)P_WHH  * 512, 4, 384, 128);

  k_encoder<<<NS / 4, 128, 0, stream>>>(cf, pad, g, pk, bp, bqkv, bo,
                                        ln1w, ln1b, bf1, bf2, ln2w, ln2b, pool);
  k_gin<<<(NS * 32 + 255) / 256, 256, 0, stream>>>(pool, g, gw, gbi, gin16);
  k_gx<<<(NS / 16) * 24 / 8, 256, 0, stream>>>(gin16, pk, bih, gx);
  k_means<<<(NS + 255) / 256, 256, 0, stream>>>(cf, mc, mo);
  k_gru<<<1, 1024, 0, stream>>>(gx, pk, bhh, hseq);
  k_heads<<<NS / 64, 128, 0, stream>>>(hseq, pool, g, mc, mo, pk,
                                       ab1, aw2, ab2, vb1, vw2, vb2,
                                       lb1, lw2, lb2, out);
  k_rollout<<<1, 128, 0, stream>>>(g, xy0, out);
}
